// GraphNetBlock_24945170055801
// MI455X (gfx1250) — matrix operand order repaired and checked by reading, not compiled, not run on hardware
//
#include <hip/hip_runtime.h>
#include <hip/hip_bf16.h>
#include <stdint.h>

// ---------------------------------------------------------------------------
// GraphNet block for MI455X (gfx1250).
// Weight-stationary (layer-1) bf16 WMMA; cooperative LDS staging; 3-deep
// A-fragment pipeline; layer-2 weights loaded per half (L2-hot) to keep the
// whole-kernel live set small; base+immediate epilogue addressing (no spills).
// ---------------------------------------------------------------------------

typedef __attribute__((ext_vector_type(16))) __bf16 v16bf;
typedef __attribute__((ext_vector_type(8)))  __bf16 v8bf;
typedef __attribute__((ext_vector_type(8)))  float  v8f;

#define DFEAT   128
#define HIDF    128
#define NNODES  50000
#define NEDGES  800000
#define ETILES  (NEDGES / 128)          // 6250
#define EGRID   1250                    // 5 edge tiles per block
#define NTILES  ((NNODES + 127) / 128)  // 391

// ---- operand builders for CDNA5 WMMA; A and B both use contiguous k = 16*(l>>4) + s -------

// A (16x32 bf16) from LDS: lane l holds row M=l&15; K = 2*kb..2*kb+15,
// kb=(l>>4)*8.  Two ds_load_b128.
__device__ __forceinline__ v16bf a_from_lds(const __bf16* p, int kb) {
  v8bf lo = *reinterpret_cast<const v8bf*>(p + 2 * kb);
  v8bf hi = *reinterpret_cast<const v8bf*>(p + 2 * kb + 8);
  v16bf a;
#pragma unroll
  for (int i = 0; i < 8; ++i) { a[i] = lo[i]; a[8 + i] = hi[i]; }
  return a;
}

// pack 8 f32 -> v8bf (v_cvt_pk_bf16_f32 x4)
__device__ __forceinline__ v8bf pack8(float4 x0, float4 x1) {
  v8bf h;
  h[0] = (__bf16)x0.x; h[1] = (__bf16)x0.y; h[2] = (__bf16)x0.z; h[3] = (__bf16)x0.w;
  h[4] = (__bf16)x1.x; h[5] = (__bf16)x1.y; h[6] = (__bf16)x1.z; h[7] = (__bf16)x1.w;
  return h;
}

// stage one 128-f32 segment -> bf16 LDS segment (unroll capped: bounded temps)
__device__ __forceinline__ void stage_seg(const float* __restrict__ src, __bf16* dst) {
#pragma unroll 4
  for (int j = 0; j < 16; ++j) {
    float4 x0 = reinterpret_cast<const float4*>(src)[2 * j];
    float4 x1 = reinterpret_cast<const float4*>(src)[2 * j + 1];
    *reinterpret_cast<v8bf*>(dst + j * 8) = pack8(x0, x1);
  }
}

// ---- weight prep: f32 [K][N] -> bf16 transposed [N][K] --------------------
__global__ void transpose_bf16_kernel(const float* __restrict__ src,
                                      __bf16* __restrict__ dst, int K, int N) {
  int i = blockIdx.x * blockDim.x + threadIdx.x;
  if (i < K * N) {
    int k = i / N, n = i % N;
    dst[n * K + k] = (__bf16)src[i];
  }
}

// Software-pipelined GEMM slice: D[4 tiles] += A(LDS) x Wreg, A prefetch
// depth 3.  NK = k-steps, base = LDS row-major bf16, ldk = row stride.
template <int NK>
__device__ __forceinline__ void gemm_pipe(const __bf16* base, int ldk,
                                          int rbase, int mrow, int kb,
                                          const v16bf* w, v8f* acc) {
  auto loadA = [&](int q) -> v16bf {
    const int ks = q >> 2, rt = q & 3;
    return a_from_lds(base + ((rbase + rt) * 16 + mrow) * ldk + ks * 32, kb);
  };
  v16bf ap[3];
  ap[0] = loadA(0);
  ap[1] = loadA(1);
  ap[2] = loadA(2);
#pragma unroll
  for (int q = 0; q < NK * 4; ++q) {
    const int rt = q & 3, ks = q >> 2;
    acc[rt] = __builtin_amdgcn_wmma_f32_16x16x32_bf16(
        false, ap[q % 3], false, w[ks], (short)0, acc[rt], false, false);
    if (q + 3 < NK * 4) ap[q % 3] = loadA(q + 3);
  }
}

// ---- edge kernel: 128 edges/tile, grid-stride -----------------------------
__global__ __launch_bounds__(256, 4) void edge_kernel(
    const float* __restrict__ nf, const float* __restrict__ ef,
    const int* __restrict__ snd, const int* __restrict__ rcv,
    const __bf16* __restrict__ W1t, const float* __restrict__ b1,
    const __bf16* __restrict__ W2t, const float* __restrict__ b2,
    float* __restrict__ agg, float* __restrict__ out_edges) {
  __shared__ __bf16 a_in[128 * 384];   // staged concat input (96 KB)
  __shared__ __bf16 act[128 * HIDF];   // layer-1 activations (32 KB)
  __shared__ int    s_rcv[128];

  const int tid   = threadIdx.x;
  const int lane  = tid & 31;
  const int wave  = tid >> 5;

  const int mrow  = lane & 15;        // A row within 16-row tile
  const int kb    = (lane >> 4) * 8;  // A K sub-offset
  const int khalf = (lane >> 4) * 16; // B K sub-offset
  const int nloc  = lane & 15;        // C column within 16-col tile
  const int mbase = (lane >> 4) * 8;  // C row base
  const int n     = wave * 16 + nloc; // this wave's output column

  // resident layer-1 weight slice for this wave's 16 output columns
  v16bf w1[12];
  {
    const __bf16* wr1 = W1t + (size_t)n * 384;
#pragma unroll
    for (int ks = 0; ks < 12; ++ks)
      w1[ks] = *reinterpret_cast<const v16bf*>(wr1 + ks * 32 + khalf);
  }
  const __bf16* wr2 = W2t + (size_t)n * HIDF + khalf;  // layer-2 row base
  const float b1v = b1[n];
  const float b2v = b2[n];

  const int srow = tid & 127;         // staging row this thread serves
  const int part = tid >> 7;          // 0: sender feats, 1: receiver feats

  for (int tile = blockIdx.x; tile < ETILES; tile += gridDim.x) {
    const int ebase = tile * 128;

    __syncthreads();   // previous tile fully consumed before restaging

    // ---- cooperative staging: concat(nf[snd], nf[rcv], ef) -> bf16 LDS ---
    {
      const int gidx = (part == 0) ? snd[ebase + srow] : rcv[ebase + srow];
      stage_seg(nf + (size_t)gidx * DFEAT, a_in + srow * 384 + part * DFEAT);
      if (tid < 128) {
        stage_seg(ef + (size_t)(ebase + tid) * DFEAT, a_in + tid * 384 + 256);
        s_rcv[tid] = rcv[ebase + tid];
      }
      const int ntile = tile + gridDim.x;   // warm L2 for next tile
      if (ntile < ETILES)
        __builtin_prefetch(ef + (size_t)ntile * 128 * DFEAT + tid * 64, 0, 1);
    }
    __syncthreads();

    // per-tile output bases: lane terms folded in once, elements use
    // compile-time offsets (24-bit IOFFSET) -> no hoistable address temps
    float*       oeb = out_edges + (size_t)ebase * DFEAT + mbase * DFEAT + n;
    const float* erb = ef        + (size_t)ebase * DFEAT + mbase * DFEAT + n;

    // ---- two half-passes of 64 rows each ---------------------------------
#pragma unroll
    for (int rh = 0; rh < 2; ++rh) {
      // layer 1: [64x384] x [384x16], B resident in VGPRs
      v8f acc[4];
#pragma unroll
      for (int rt = 0; rt < 4; ++rt)
#pragma unroll
        for (int i = 0; i < 8; ++i) acc[rt][i] = 0.0f;

      gemm_pipe<12>(a_in, 384, rh * 4, mrow, kb, w1, acc);

      // bias + ReLU -> act rows of this half (16-col slice per wave)
#pragma unroll
      for (int rt = 0; rt < 4; ++rt)
#pragma unroll
        for (int i = 0; i < 8; ++i) {
          float v = acc[rt][i] + b1v;
          v = v > 0.0f ? v : 0.0f;
          act[((rh * 4 + rt) * 16 + mbase + i) * HIDF + n] = (__bf16)v;
        }
      __syncthreads();

      // layer 2: [64x128] x [128x16]; B fragments loaded per half (L2-hot),
      // short-lived so they don't inflate the whole-kernel live set
      v16bf w2l[4];
#pragma unroll
      for (int ks = 0; ks < 4; ++ks)
        w2l[ks] = *reinterpret_cast<const v16bf*>(wr2 + ks * 32);

      v8f acc2[4];
#pragma unroll
      for (int rt = 0; rt < 4; ++rt)
#pragma unroll
        for (int i = 0; i < 8; ++i) acc2[rt][i] = 0.0f;

      gemm_pipe<4>(act, HIDF, rh * 4, mrow, kb, w2l, acc2);

      // epilogue: residual store + f32 atomic scatter into agg
#pragma unroll
      for (int rt = 0; rt < 4; ++rt)
#pragma unroll
        for (int i = 0; i < 8; ++i) {
          const int cm = (rh * 4 + rt) * 16 + i;       // compile-time part
          const float val = acc2[rt][i] + b2v;         // updated_edges
          atomicAdd(agg + (size_t)s_rcv[cm + mbase] * DFEAT + n, val);
          oeb[cm * DFEAT] = val + erb[cm * DFEAT];
        }
    }
  }
}

// ---- node kernel: 128 nodes/tile ------------------------------------------
__global__ __launch_bounds__(256, 4) void node_kernel(
    const float* __restrict__ nf, const float* __restrict__ agg,
    const __bf16* __restrict__ W1t, const float* __restrict__ b1,
    const __bf16* __restrict__ W2t, const float* __restrict__ b2,
    float* __restrict__ out_nodes) {
  __shared__ __bf16 a_in[128 * 256];   // staged concat(node, agg) (64 KB)
  __shared__ __bf16 act[128 * HIDF];   // layer-1 activations     (32 KB)

  const int tid   = threadIdx.x;
  const int lane  = tid & 31;
  const int wave  = tid >> 5;

  const int mrow  = lane & 15;
  const int kb    = (lane >> 4) * 8;
  const int khalf = (lane >> 4) * 16;
  const int nloc  = lane & 15;
  const int mbase = (lane >> 4) * 8;
  const int n     = wave * 16 + nloc;

  v16bf w1[8];
  {
    const __bf16* wr1 = W1t + (size_t)n * 256;
#pragma unroll
    for (int ks = 0; ks < 8; ++ks)
      w1[ks] = *reinterpret_cast<const v16bf*>(wr1 + ks * 32 + khalf);
  }
  const __bf16* wr2 = W2t + (size_t)n * HIDF + khalf;
  const float b1v = b1[n];
  const float b2v = b2[n];

  const int srow = tid & 127;
  const int part = tid >> 7;          // 0: node feats, 1: aggregated edges

  for (int tile = blockIdx.x; tile < NTILES; tile += gridDim.x) {
    const int nbase = tile * 128;

    __syncthreads();

    // ---- cooperative staging: concat(nf, agg) -> bf16 LDS ----------------
    {
      const int nd_raw = nbase + srow;
      const int nd = nd_raw < NNODES ? nd_raw : NNODES - 1;   // clamp loads
      const float* src = (part == 0) ? nf  + (size_t)nd * DFEAT
                                     : agg + (size_t)nd * DFEAT;
      stage_seg(src, a_in + srow * 256 + part * DFEAT);
    }
    __syncthreads();

    float*       onb = out_nodes + (size_t)nbase * DFEAT + mbase * DFEAT + n;
    const float* nrb = nf        + (size_t)nbase * DFEAT + mbase * DFEAT + n;

#pragma unroll
    for (int rh = 0; rh < 2; ++rh) {
      v8f acc[4];
#pragma unroll
      for (int rt = 0; rt < 4; ++rt)
#pragma unroll
        for (int i = 0; i < 8; ++i) acc[rt][i] = 0.0f;

      gemm_pipe<8>(a_in, 256, rh * 4, mrow, kb, w1, acc);

#pragma unroll
      for (int rt = 0; rt < 4; ++rt)
#pragma unroll
        for (int i = 0; i < 8; ++i) {
          float v = acc[rt][i] + b1v;
          v = v > 0.0f ? v : 0.0f;
          act[((rh * 4 + rt) * 16 + mbase + i) * HIDF + n] = (__bf16)v;
        }
      __syncthreads();

      v16bf w2l[4];
#pragma unroll
      for (int ks = 0; ks < 4; ++ks)
        w2l[ks] = *reinterpret_cast<const v16bf*>(wr2 + ks * 32);

      v8f acc2[4];
#pragma unroll
      for (int rt = 0; rt < 4; ++rt)
#pragma unroll
        for (int i = 0; i < 8; ++i) acc2[rt][i] = 0.0f;

      gemm_pipe<4>(act, HIDF, rh * 4, mrow, kb, w2l, acc2);

#pragma unroll
      for (int rt = 0; rt < 4; ++rt)
#pragma unroll
        for (int i = 0; i < 8; ++i) {
          const int cm = (rh * 4 + rt) * 16 + i;
          const int nd = nbase + cm + mbase;
          if (nd < NNODES) {
            const float val = acc2[rt][i] + b2v;
            onb[cm * DFEAT] = val + nrb[cm * DFEAT];
          }
        }
    }
  }
}

// ---------------------------------------------------------------------------
extern "C" void kernel_launch(void* const* d_in, const int* in_sizes, int n_in,
                              void* d_out, int out_size, void* d_ws, size_t ws_size,
                              hipStream_t stream) {
  const float* nf  = (const float*)d_in[0];
  const float* ef  = (const float*)d_in[1];
  const int*   snd = (const int*)d_in[2];
  const int*   rcv = (const int*)d_in[3];
  const float* We1 = (const float*)d_in[4];
  const float* be1 = (const float*)d_in[5];
  const float* We2 = (const float*)d_in[6];
  const float* be2 = (const float*)d_in[7];
  const float* Wn1 = (const float*)d_in[8];
  const float* bn1 = (const float*)d_in[9];
  const float* Wn2 = (const float*)d_in[10];
  const float* bn2 = (const float*)d_in[11];

  // workspace layout
  char*   ws   = (char*)d_ws;
  __bf16* We1t = (__bf16*)(ws);                              // 128 x 384
  __bf16* We2t = (__bf16*)(ws + 98304);                      // 128 x 128
  __bf16* Wn1t = (__bf16*)(ws + 98304 + 32768);              // 128 x 256
  __bf16* Wn2t = (__bf16*)(ws + 98304 + 32768 + 65536);      // 128 x 128
  float*  agg  = (float*)(ws + 262144);                      // 50000 x 128 f32

  float* out_nodes = (float*)d_out;
  float* out_edges = (float*)d_out + (size_t)NNODES * DFEAT;

  transpose_bf16_kernel<<<(384 * 128 + 255) / 256, 256, 0, stream>>>(We1, We1t, 384, 128);
  transpose_bf16_kernel<<<(128 * 128 + 255) / 256, 256, 0, stream>>>(We2, We2t, 128, 128);
  transpose_bf16_kernel<<<(256 * 128 + 255) / 256, 256, 0, stream>>>(Wn1, Wn1t, 256, 128);
  transpose_bf16_kernel<<<(128 * 128 + 255) / 256, 256, 0, stream>>>(Wn2, Wn2t, 128, 128);

  hipMemsetAsync(agg, 0, (size_t)NNODES * DFEAT * sizeof(float), stream);

  edge_kernel<<<EGRID, 256, 0, stream>>>(nf, ef, snd, rcv, We1t, be1,
                                         We2t, be2, agg, out_edges);
  node_kernel<<<NTILES, 256, 0, stream>>>(nf, agg, Wn1t, bn1,
                                          Wn2t, bn2, out_nodes);
}
